// HGTModel_24739011625780
// MI455X (gfx1250) — compile-verified, matrix operand order checked by reading, not run on hardware
//
#include <hip/hip_runtime.h>
#include <hip/hip_bf16.h>
#include <math.h>

// ---------------------------------------------------------------------------
// HGT (heterogeneous graph transformer) for MI455X / gfx1250.
//  * All dense GEMMs -> v_wmma_f32_16x16x32_bf16, both operands bf16 in memory
//    (inner loop = pure b128 loads + WMMA, no conversion VALU).
//  * Relation transforms hoisted from per-edge to per-node via block-diagonal
//    GEMMs (mathematically identical, 12-25x fewer FLOPs).
//  * Edge segment-softmax via 3 scatter passes with f32 atomics.
//  * fp32 kept for softmax / layernorm / residual / atomic accumulation.
// ---------------------------------------------------------------------------

typedef __attribute__((ext_vector_type(16))) __bf16 v16bf;
typedef __attribute__((ext_vector_type(8)))  __bf16 v8bf;
typedef __attribute__((ext_vector_type(4)))  __bf16 v4bf;
typedef __attribute__((ext_vector_type(8)))  float  v8f;

#define HID   128
#define HEADS 4
#define DH    32
#define NNEWS 40000
#define NENT  20000
#define FNEWS 768
#define FENT  384
#define NEDGE 512000

__device__ __forceinline__ __bf16 f2bf(float f) {
  unsigned u = __float_as_uint(f);
  unsigned r = u + 0x7FFFu + ((u >> 16) & 1u);   // round-to-nearest-even
  unsigned short h = (unsigned short)(r >> 16);
  return __builtin_bit_cast(__bf16, h);
}

__device__ __forceinline__ void atomicMaxFloat(float* addr, float v) {
  unsigned* ua = (unsigned*)addr;
  unsigned old = *ua;
  for (;;) {
    float f = __uint_as_float(old);
    if (f >= v) break;
    unsigned prev = atomicCAS(ua, old, __float_as_uint(v));
    if (prev == old) break;
    old = prev;
  }
}

// ------------------------------- utility -----------------------------------

__global__ void k_fill(float* __restrict__ p, float v, long n) {
  long i = (long)blockIdx.x * blockDim.x + threadIdx.x;
  if (i < n) p[i] = v;
}

// bulk fp32 -> bf16 (4 elems / thread)
__global__ void k_f2bf(const float* __restrict__ in, __bf16* __restrict__ out,
                       long n4) {
  long i = (long)blockIdx.x * blockDim.x + threadIdx.x;
  if (i >= n4) return;
  float4 v = ((const float4*)in)[i];
  v4bf o;
  o[0] = f2bf(v.x); o[1] = f2bf(v.y); o[2] = f2bf(v.z); o[3] = f2bf(v.w);
  ((v4bf*)out)[i] = o;
}

// W [K x 128] fp32  ->  Wt [128 x K] bf16   (transposed so GEMM B loads are
// contiguous along K: one 16B load per half-fragment per lane)
__global__ void k_wprep(const float* __restrict__ W, __bf16* __restrict__ Wt, int K) {
  int t = blockIdx.x * blockDim.x + threadIdx.x;
  if (t >= K * HID) return;
  int n = t / K, k = t - n * K;
  Wt[t] = f2bf(W[(long)k * HID + n]);
}

// a_rel/m_rel [4 x 32 x 32] -> block-diagonal 128x128, transposed bf16 [N x K]
__global__ void k_bdprep(const float* __restrict__ A, __bf16* __restrict__ Bt) {
  int t = blockIdx.x * blockDim.x + threadIdx.x;
  if (t >= HID * HID) return;
  int n = t >> 7, k = t & 127;
  int h = n >> 5;
  float v = ((k >> 5) == h) ? A[h * DH * DH + (k & 31) * DH + (n & 31)] : 0.f;
  Bt[t] = f2bf(v);
}

// ------------------------------- GEMM ---------------------------------------
// C[M x 128] = A[M x K] (bf16) @ B (bf16, stored transposed Bt[128 x K]) + bias
// Block = 256 threads (8 waves). Block owns a 16-row strip; wave w owns the
// 16-column tile n0 = 16*w. K stepped by 32 with v_wmma_f32_16x16x32_bf16.
// Fragment layout per CDNA5 ISA 7.12.2: lane l (r=l&15, half=l>>4) holds
// A row m0+r / B col n0+r, elements 0..7 = K[k0..k0+7], 8..15 = K[k0+16..+23]
// with k0 = kk + 8*half. C: VGPR j -> row m0 + 8*half + j, col n0 + r.
// Output written as fp32 (Cf) and/or bf16 (Cb) -- whichever is non-null.
__global__ __launch_bounds__(256)
void k_gemm(const __bf16* __restrict__ A, const __bf16* __restrict__ Bt,
            const float* __restrict__ bias,
            float* __restrict__ Cf, __bf16* __restrict__ Cb,
            int M, int K) {
  const int l    = threadIdx.x & 31;
  const int wv   = threadIdx.x >> 5;
  const int r    = l & 15;
  const int half = l >> 4;
  const int m0   = blockIdx.x << 4;
  const int n0   = wv << 4;
  if (m0 >= M) return;

  const __bf16* Ar = A  + (size_t)(m0 + r) * K;
  const __bf16* Br = Bt + (size_t)(n0 + r) * K;

  v8f acc = {0.f, 0.f, 0.f, 0.f, 0.f, 0.f, 0.f, 0.f};

  for (int kk = 0; kk < K; kk += 32) {
    const int k0 = kk + (half << 3);
    const int k1 = k0 + 16;

    v8bf a0 = *(const v8bf*)(Ar + k0);
    v8bf a1 = *(const v8bf*)(Ar + k1);
    v8bf b0 = *(const v8bf*)(Br + k0);
    v8bf b1 = *(const v8bf*)(Br + k1);
    v16bf av, bv;
#pragma unroll
    for (int i = 0; i < 8; ++i) {
      av[i] = a0[i]; av[8 + i] = a1[i];
      bv[i] = b0[i]; bv[8 + i] = b1[i];
    }

    acc = __builtin_amdgcn_wmma_f32_16x16x32_bf16(
        /*neg_a=*/false, av, /*neg_b=*/false, bv,
        /*c_mod=*/(short)0, acc, /*reuse_a=*/false, /*reuse_b=*/false);
  }

  const int col  = n0 + r;
  const float bs = bias ? bias[col] : 0.f;
  const int rowb = m0 + (half << 3);
  float res[8];
#pragma unroll
  for (int j = 0; j < 8; ++j) res[j] = acc[j] + bs;
  if (Cf) {
#pragma unroll
    for (int j = 0; j < 8; ++j)
      Cf[(size_t)(rowb + j) * HID + col] = res[j];
  }
  if (Cb) {
#pragma unroll
    for (int j = 0; j < 8; ++j)
      Cb[(size_t)(rowb + j) * HID + col] = f2bf(res[j]);
  }
}

// ------------------------- layernorm (+residual) + relu ---------------------
// One wave32 per row of 128; lane handles 4 contiguous floats.
// o == nullptr : x = relu(LN(h))
// else         : x = relu(LN(h + s*o + (1-s)*h)),  s = sigmoid(skip)
// Writes fp32 x and (if xb) a fused bf16 copy for the following GEMMs.
__global__ __launch_bounds__(256)
void k_ln_relu(const float* __restrict__ h, const float* __restrict__ o,
               const float* __restrict__ skipp,
               const float* __restrict__ g, const float* __restrict__ b,
               float* __restrict__ x, __bf16* __restrict__ xb, int M) {
  int row = blockIdx.x * 8 + (threadIdx.x >> 5);
  int l   = threadIdx.x & 31;
  if (row >= M) return;

  float4 v = ((const float4*)(h + (size_t)row * HID))[l];
  if (o) {
    float sg = 1.f / (1.f + __expf(-skipp[0]));
    float4 ov = ((const float4*)(o + (size_t)row * HID))[l];
    v.x = (2.f - sg) * v.x + sg * ov.x;
    v.y = (2.f - sg) * v.y + sg * ov.y;
    v.z = (2.f - sg) * v.z + sg * ov.z;
    v.w = (2.f - sg) * v.w + sg * ov.w;
  }
  float sum = v.x + v.y + v.z + v.w;
#pragma unroll
  for (int m = 16; m >= 1; m >>= 1) sum += __shfl_xor(sum, m, 32);
  float mu = sum * (1.f / HID);
  float dx = v.x - mu, dy = v.y - mu, dz = v.z - mu, dw = v.w - mu;
  float vs = dx * dx + dy * dy + dz * dz + dw * dw;
#pragma unroll
  for (int m = 16; m >= 1; m >>= 1) vs += __shfl_xor(vs, m, 32);
  float rs = rsqrtf(vs * (1.f / HID) + 1e-5f);

  float4 g4 = ((const float4*)g)[l];
  float4 b4 = ((const float4*)b)[l];
  float4 y;
  y.x = fmaxf(dx * rs * g4.x + b4.x, 0.f);
  y.y = fmaxf(dy * rs * g4.y + b4.y, 0.f);
  y.z = fmaxf(dz * rs * g4.z + b4.z, 0.f);
  y.w = fmaxf(dw * rs * g4.w + b4.w, 0.f);
  ((float4*)(x + (size_t)row * HID))[l] = y;
  if (xb) {
    v4bf yb;
    yb[0] = f2bf(y.x); yb[1] = f2bf(y.y); yb[2] = f2bf(y.z); yb[3] = f2bf(y.w);
    ((v4bf*)(xb + (size_t)row * HID))[l] = yb;
  }
}

// ------------------------------- edge phase ---------------------------------

// pass 1: alpha[e,h] = (q[dst,h,:] . kA[src,h,:]) * p[h] / sqrt(32); seg max
__global__ __launch_bounds__(256)
void k_edge_alpha(const int* __restrict__ ei, int E,
                  const float* __restrict__ ka, const float* __restrict__ q,
                  const float* __restrict__ prel,
                  float* __restrict__ al, float* __restrict__ mmax) {
  int t = blockIdx.x * blockDim.x + threadIdx.x;
  if (t >= E * HEADS) return;
  int e = t >> 2, hh = t & 3;
  int s = ei[e], d = ei[E + e];
  const float4* kp = (const float4*)(ka + (size_t)s * HID + hh * DH);
  const float4* qp = (const float4*)(q  + (size_t)d * HID + hh * DH);
  float acc = 0.f;
#pragma unroll
  for (int i = 0; i < 8; ++i) {
    float4 a = kp[i], bq = qp[i];
    acc += a.x * bq.x + a.y * bq.y + a.z * bq.z + a.w * bq.w;
  }
  float av = acc * prel[hh] * 0.17677669529663689f;  // 1/sqrt(32)
  al[t] = av;
  atomicMaxFloat(&mmax[(size_t)d * HEADS + hh], av);
}

// pass 2: e = exp(alpha - max[dst]); seg sum
__global__ __launch_bounds__(256)
void k_edge_exp(const int* __restrict__ ei, int E,
                float* __restrict__ al, const float* __restrict__ mmax,
                float* __restrict__ ssum) {
  int t = blockIdx.x * blockDim.x + threadIdx.x;
  if (t >= E * HEADS) return;
  int e = t >> 2, hh = t & 3;
  int d = ei[E + e];
  float ev = __expf(al[t] - mmax[(size_t)d * HEADS + hh]);
  al[t] = ev;
  atomicAdd(&ssum[(size_t)d * HEADS + hh], ev);
}

// pass 3: agg[dst,c] += (e / sum[dst]) * vM[src,c]; thread per (edge, channel)
__global__ __launch_bounds__(256)
void k_edge_agg(const int* __restrict__ ei, int E,
                const float* __restrict__ al, const float* __restrict__ ssum,
                const float* __restrict__ vm, float* __restrict__ agg) {
  int t = blockIdx.x * blockDim.x + threadIdx.x;
  if (t >= E * HID) return;
  int e = t >> 7, c = t & 127;
  int hh = c >> 5;
  int s = ei[e], d = ei[E + e];
  float w = al[(size_t)e * HEADS + hh] /
            (ssum[(size_t)d * HEADS + hh] + 1e-16f);
  atomicAdd(&agg[(size_t)d * HID + c], w * vm[(size_t)s * HID + c]);
}

// exact gelu, fp32 in -> bf16 out (feeds the O-projection GEMM)
__global__ void k_gelu_bf(const float* __restrict__ in, __bf16* __restrict__ out,
                          long n4) {
  long i = (long)blockIdx.x * blockDim.x + threadIdx.x;
  if (i >= n4) return;
  float4 v = ((const float4*)in)[i];
  v4bf o;
  o[0] = f2bf(0.5f * v.x * (1.f + erff(v.x * 0.70710678118654752f)));
  o[1] = f2bf(0.5f * v.y * (1.f + erff(v.y * 0.70710678118654752f)));
  o[2] = f2bf(0.5f * v.z * (1.f + erff(v.z * 0.70710678118654752f)));
  o[3] = f2bf(0.5f * v.w * (1.f + erff(v.w * 0.70710678118654752f)));
  ((v4bf*)out)[i] = o;
}

// out[M x 2] = x[M x 128] @ w[128 x 2] + b ; wave per row
__global__ __launch_bounds__(256)
void k_out(const float* __restrict__ x, const float* __restrict__ w,
           const float* __restrict__ b, float* __restrict__ out, int M) {
  int row = blockIdx.x * 8 + (threadIdx.x >> 5);
  int l   = threadIdx.x & 31;
  if (row >= M) return;
  float4 xv = ((const float4*)(x + (size_t)row * HID))[l];
  const float* wp = w + (size_t)(4 * l) * 2;
  float a0 = xv.x * wp[0] + xv.y * wp[2] + xv.z * wp[4] + xv.w * wp[6];
  float a1 = xv.x * wp[1] + xv.y * wp[3] + xv.z * wp[5] + xv.w * wp[7];
#pragma unroll
  for (int m = 16; m >= 1; m >>= 1) {
    a0 += __shfl_xor(a0, m, 32);
    a1 += __shfl_xor(a1, m, 32);
  }
  if (l == 0) {
    out[(size_t)row * 2 + 0] = a0 + b[0];
    out[(size_t)row * 2 + 1] = a1 + b[1];
  }
}

// ------------------------------- host side ----------------------------------
// d_in flattening (insertion order, recursive):
//   0 x_news  1 x_entity
//   2/3 in_lin.news.{w,b}  4/5 in_lin.entity.{w,b}
//   6/7 norm.news.{g,b}    8/9 norm.entity.{g,b}
//   per layer L (base LB = 10 + 24*L):
//     +0..+5  kqv.news k{w,b} q{w,b} v{w,b}
//     +6..+11 kqv.entity
//     +12/13  o.news {w,b}  +14/15 o.entity {w,b}
//     +16/17  skip news/entity
//     +18/19  a_rel has/in  +20/21 m_rel has/in  +22/23 p_rel has/in
//   58/59 out_lin {w,b}    60 ei_ne   61 ei_en

extern "C" void kernel_launch(void* const* d_in, const int* in_sizes, int n_in,
                              void* d_out, int out_size, void* d_ws, size_t ws_size,
                              hipStream_t stream) {
  (void)in_sizes; (void)n_in; (void)out_size; (void)ws_size;
  auto P = [&](int i) { return (const float*)d_in[i]; };
  const int* ei_ne = (const int*)d_in[60];
  const int* ei_en = (const int*)d_in[61];

  // ---- workspace carve-out (256B aligned) ----
  char* base = (char*)d_ws;
  size_t cur = 0;
  auto alloc = [&](size_t bytes) -> char* {
    char* p = base + cur;
    cur = (cur + bytes + 255) & ~(size_t)255;
    return p;
  };
  const size_t NB  = (size_t)NNEWS * HID * sizeof(float);
  const size_t EB  = (size_t)NENT  * HID * sizeof(float);
  const size_t NBH = (size_t)NNEWS * HID * sizeof(__bf16);
  const size_t EBH = (size_t)NENT  * HID * sizeof(__bf16);
  // fp32
  float* xn   = (float*)alloc(NB);
  float* qn   = (float*)alloc(NB);
  float* kan  = (float*)alloc(NB);
  float* vmn  = (float*)alloc(NB);
  float* aggn = (float*)alloc(NB);   // also input-stage pre-LN temp
  float* on   = (float*)alloc(NB);
  float* xe   = (float*)alloc(EB);
  float* qe   = (float*)alloc(EB);
  float* kae  = (float*)alloc(EB);
  float* vme  = (float*)alloc(EB);
  float* agge = (float*)alloc(EB);
  float* oe   = (float*)alloc(EB);
  // bf16 activations
  __bf16* xnb = (__bf16*)alloc(NBH);
  __bf16* knb = (__bf16*)alloc(NBH);
  __bf16* vnb = (__bf16*)alloc(NBH);
  __bf16* gnb = (__bf16*)alloc(NBH);
  __bf16* xeb = (__bf16*)alloc(EBH);
  __bf16* keb = (__bf16*)alloc(EBH);
  __bf16* veb = (__bf16*)alloc(EBH);
  __bf16* geb = (__bf16*)alloc(EBH);
  __bf16* inb_n = (__bf16*)alloc((size_t)NNEWS * FNEWS * sizeof(__bf16));
  __bf16* inb_e = (__bf16*)alloc((size_t)NENT  * FENT  * sizeof(__bf16));
  // softmax state
  float* al = (float*)alloc((size_t)NEDGE * HEADS * sizeof(float));
  float* mn = (float*)alloc((size_t)NNEWS * HEADS * sizeof(float));
  float* sn = (float*)alloc((size_t)NNEWS * HEADS * sizeof(float));
  float* me = (float*)alloc((size_t)NENT  * HEADS * sizeof(float));
  float* se = (float*)alloc((size_t)NENT  * HEADS * sizeof(float));
  // weights (bf16, transposed)
  __bf16* wtin_n = (__bf16*)alloc((size_t)HID * FNEWS * sizeof(__bf16));
  __bf16* wtin_e = (__bf16*)alloc((size_t)HID * FENT  * sizeof(__bf16));
  __bf16* wt[2][2][4];   // [layer][0=news,1=entity][k,q,v,o]
  for (int L = 0; L < 2; ++L)
    for (int T = 0; T < 2; ++T)
      for (int j = 0; j < 4; ++j)
        wt[L][T][j] = (__bf16*)alloc((size_t)HID * HID * sizeof(__bf16));
  __bf16* bd[2][4];      // [layer][a_has, a_in, m_has, m_in]
  for (int L = 0; L < 2; ++L)
    for (int j = 0; j < 4; ++j)
      bd[L][j] = (__bf16*)alloc((size_t)HID * HID * sizeof(__bf16));

  // ---- launch helpers ----
  auto wprep = [&](const float* W, __bf16* Wt, int K) {
    int total = K * HID;
    k_wprep<<<(total + 255) / 256, 256, 0, stream>>>(W, Wt, K);
  };
  auto bdprep = [&](const float* A, __bf16* Bt) {
    k_bdprep<<<(HID * HID + 255) / 256, 256, 0, stream>>>(A, Bt);
  };
  auto gemm = [&](const __bf16* A, const __bf16* Bt, const float* bias,
                  float* Cf, __bf16* Cb, int M, int K) {
    k_gemm<<<M / 16, 256, 0, stream>>>(A, Bt, bias, Cf, Cb, M, K);
  };
  auto ln = [&](const float* h, const float* o, const float* sk,
                const float* g, const float* b, float* x, __bf16* xb, int M) {
    k_ln_relu<<<(M + 7) / 8, 256, 0, stream>>>(h, o, sk, g, b, x, xb, M);
  };
  auto fill = [&](float* p, float v, long n) {
    k_fill<<<(int)((n + 255) / 256), 256, 0, stream>>>(p, v, n);
  };
  auto cvt = [&](const float* in, __bf16* out, long n) {
    k_f2bf<<<(int)((n / 4 + 255) / 256), 256, 0, stream>>>(in, out, n / 4);
  };

  // ---- weight prep (bf16, transposed / block-diagonalized) ----
  wprep(P(2), wtin_n, FNEWS);
  wprep(P(4), wtin_e, FENT);
  for (int L = 0; L < 2; ++L) {
    int LB = 10 + 24 * L;
    wprep(P(LB + 0),  wt[L][0][0], HID);  // news  k
    wprep(P(LB + 2),  wt[L][0][1], HID);  // news  q
    wprep(P(LB + 4),  wt[L][0][2], HID);  // news  v
    wprep(P(LB + 12), wt[L][0][3], HID);  // news  o
    wprep(P(LB + 6),  wt[L][1][0], HID);  // ent   k
    wprep(P(LB + 8),  wt[L][1][1], HID);  // ent   q
    wprep(P(LB + 10), wt[L][1][2], HID);  // ent   v
    wprep(P(LB + 14), wt[L][1][3], HID);  // ent   o
    bdprep(P(LB + 18), bd[L][0]);         // a_rel has
    bdprep(P(LB + 19), bd[L][1]);         // a_rel in
    bdprep(P(LB + 20), bd[L][2]);         // m_rel has
    bdprep(P(LB + 21), bd[L][3]);         // m_rel in
  }

  // ---- input projections + LN + relu ----
  cvt(P(0), inb_n, (long)NNEWS * FNEWS);
  cvt(P(1), inb_e, (long)NENT * FENT);
  gemm(inb_n, wtin_n, P(3), aggn, nullptr, NNEWS, FNEWS);
  ln(aggn, nullptr, nullptr, P(6), P(7), xn, xnb, NNEWS);
  gemm(inb_e, wtin_e, P(5), agge, nullptr, NENT, FENT);
  ln(agge, nullptr, nullptr, P(8), P(9), xe, xeb, NENT);

  // ---- HGT conv layers ----
  for (int L = 0; L < 2; ++L) {
    int LB = 10 + 24 * L;
    // K/Q/V projections (k,v only needed as bf16 GEMM inputs; q fp32 for edges)
    gemm(xnb, wt[L][0][0], P(LB + 1),  nullptr, knb, NNEWS, HID);
    gemm(xnb, wt[L][0][1], P(LB + 3),  qn, nullptr,  NNEWS, HID);
    gemm(xnb, wt[L][0][2], P(LB + 5),  nullptr, vnb, NNEWS, HID);
    gemm(xeb, wt[L][1][0], P(LB + 7),  nullptr, keb, NENT, HID);
    gemm(xeb, wt[L][1][1], P(LB + 9),  qe, nullptr,  NENT, HID);
    gemm(xeb, wt[L][1][2], P(LB + 11), nullptr, veb, NENT, HID);
    // per-node relation transforms (hoisted from per-edge)
    gemm(knb, bd[L][0], nullptr, kan, nullptr, NNEWS, HID);  // news src of "has"
    gemm(vnb, bd[L][2], nullptr, vmn, nullptr, NNEWS, HID);
    gemm(keb, bd[L][1], nullptr, kae, nullptr, NENT, HID);   // ent  src of "in"
    gemm(veb, bd[L][3], nullptr, vme, nullptr, NENT, HID);
    // segment-softmax state
    fill(mn, -3.0e38f, (long)NNEWS * HEADS);
    fill(sn, 0.f,      (long)NNEWS * HEADS);
    fill(me, -3.0e38f, (long)NENT * HEADS);
    fill(se, 0.f,      (long)NENT * HEADS);
    fill(aggn, 0.f,    (long)NNEWS * HID);
    fill(agge, 0.f,    (long)NENT * HID);
    // relation "has": news -> entity
    {
      int tEH = NEDGE * HEADS, tEC = NEDGE * HID;
      k_edge_alpha<<<(tEH + 255) / 256, 256, 0, stream>>>(ei_ne, NEDGE, kan, qe, P(LB + 22), al, me);
      k_edge_exp  <<<(tEH + 255) / 256, 256, 0, stream>>>(ei_ne, NEDGE, al, me, se);
      k_edge_agg  <<<(tEC + 255) / 256, 256, 0, stream>>>(ei_ne, NEDGE, al, se, vmn, agge);
    }
    // relation "in": entity -> news
    {
      int tEH = NEDGE * HEADS, tEC = NEDGE * HID;
      k_edge_alpha<<<(tEH + 255) / 256, 256, 0, stream>>>(ei_en, NEDGE, kae, qn, P(LB + 23), al, mn);
      k_edge_exp  <<<(tEH + 255) / 256, 256, 0, stream>>>(ei_en, NEDGE, al, mn, sn);
      k_edge_agg  <<<(tEC + 255) / 256, 256, 0, stream>>>(ei_en, NEDGE, al, sn, vme, aggn);
    }
    // gelu -> O projection -> skip-gated residual + LN + relu
    k_gelu_bf<<<(int)(((long)NENT * HID / 4 + 255) / 256), 256, 0, stream>>>(agge, geb, (long)NENT * HID / 4);
    k_gelu_bf<<<(int)(((long)NNEWS * HID / 4 + 255) / 256), 256, 0, stream>>>(aggn, gnb, (long)NNEWS * HID / 4);
    gemm(gnb, wt[L][0][3], P(LB + 13), on, nullptr, NNEWS, HID);
    gemm(geb, wt[L][1][3], P(LB + 15), oe, nullptr, NENT, HID);
    ln(xn, on, P(LB + 16), P(6), P(7), xn, xnb, NNEWS);
    ln(xe, oe, P(LB + 17), P(8), P(9), xe, xeb, NENT);
  }

  // ---- output head ----
  k_out<<<(NNEWS + 7) / 8, 256, 0, stream>>>(xn, P(58), P(59), (float*)d_out, NNEWS);
}
